// SwinTransformerBlock_12068858102223
// MI455X (gfx1250) — matrix operand order repaired and checked by reading, not compiled, not run on hardware
//
#include <hip/hip_runtime.h>

typedef __attribute__((ext_vector_type(16))) __bf16 v16bf;
typedef __attribute__((ext_vector_type(8)))  __bf16 v8bf;
typedef __attribute__((ext_vector_type(8)))  float  v8f;
typedef __attribute__((ext_vector_type(4)))  int    v4i;

static constexpr int kH = 56, kW = 56, kSS = 3;
static constexpr int kHeads = 12, kC = 384, kN = 49;
static constexpr int kB = 32;
static constexpr int kM = kB * kH * kW;          // 100352 tokens / windowed rows
static constexpr int kQKV = 3 * kC;              // 1152
static constexpr int kHid = 4 * kC;              // 1536
static constexpr int kWinTotal = kB * 64;        // 2048 windows

// ----------------------------------------------------- async copy primitives
#if __has_builtin(__builtin_amdgcn_global_load_async_to_lds_b128)
#define HAVE_ASYNC_LDS 1
#else
#define HAVE_ASYNC_LDS 0
#endif

// Copy 16 bytes global -> LDS at byte offset OFF (applied to both addresses).
template <int OFF>
__device__ __forceinline__ void async_cp16(const __bf16* src, __bf16* dst) {
#if HAVE_ASYNC_LDS
  __builtin_amdgcn_global_load_async_to_lds_b128(
      (__attribute__((address_space(1))) v4i*)(void*)src,
      (__attribute__((address_space(3))) v4i*)(void*)dst, OFF, 0);
#else
  *(v8bf*)((char*)dst + OFF) = *(const v8bf*)((const char*)src + OFF);
#endif
}

__device__ __forceinline__ void wait_async_le4() {
#if HAVE_ASYNC_LDS
#if __has_builtin(__builtin_amdgcn_s_wait_asynccnt)
  __builtin_amdgcn_s_wait_asynccnt(4);
#else
  asm volatile("s_wait_asynccnt 0x4" ::: "memory");
#endif
#endif
}

__device__ __forceinline__ void wait_async_le0() {
#if HAVE_ASYNC_LDS
#if __has_builtin(__builtin_amdgcn_s_wait_asynccnt)
  __builtin_amdgcn_s_wait_asynccnt(0);
#else
  asm volatile("s_wait_asynccnt 0x0" ::: "memory");
#endif
#endif
}

// ---------------------------------------------------------------- utilities

__global__ __launch_bounds__(256) void f2bf_kernel(const float* __restrict__ in,
                                                   __bf16* __restrict__ out, int n) {
  int i = blockIdx.x * 256 + threadIdx.x;
  if (i < n) out[i] = (__bf16)in[i];
}

// bias_s[h][m][n] = rpb_table[rel_idx(m,n)][h]
__global__ __launch_bounds__(256) void rpb_expand_kernel(const float* __restrict__ rpb,
                                                         float* __restrict__ bias_s) {
  int i = blockIdx.x * 256 + threadIdx.x;
  if (i >= kHeads * kN * kN) return;
  int h = i / (kN * kN);
  int rem = i % (kN * kN);
  int m = rem / kN, n = rem % kN;
  int idx = (m / 7 - n / 7 + 6) * 13 + (m % 7 - n % 7 + 6);
  bias_s[i] = rpb[idx * kHeads + h];
}

// ------------------------------------------------- LN1 + shift + window part
// One wave per output row (win*49 + n). Source pixel = roll(-3,-3) of x.
__global__ __launch_bounds__(256)
void ln1_shift_window_kernel(const float* __restrict__ x, const float* __restrict__ gw,
                             const float* __restrict__ bw, __bf16* __restrict__ Y) {
  const int row = blockIdx.x * 8 + (threadIdx.x >> 5);
  const int lane = threadIdx.x & 31;
  const int win = row / kN, n = row % kN;
  const int b = win >> 6, wi = win & 63;
  const int wh = wi >> 3, ww = wi & 7;
  const int pi = n / 7, pj = n % 7;
  int r = wh * 7 + pi + kSS; if (r >= kH) r -= kH;
  int c = ww * 7 + pj + kSS; if (c >= kW) c -= kW;
  const float* src = x + ((size_t)b * (kH * kW) + (size_t)r * kW + c) * kC;
  float v[12];
  float s = 0.f, s2 = 0.f;
#pragma unroll
  for (int e = 0; e < 12; e++) { float t = src[lane + 32 * e]; v[e] = t; s += t; s2 += t * t; }
#pragma unroll
  for (int o = 16; o > 0; o >>= 1) { s += __shfl_xor(s, o, 32); s2 += __shfl_xor(s2, o, 32); }
  const float mean = s * (1.f / (float)kC);
  const float var  = s2 * (1.f / (float)kC) - mean * mean;
  const float rstd = rsqrtf(var + 1e-5f);
  __bf16* dst = Y + (size_t)row * kC;
#pragma unroll
  for (int e = 0; e < 12; e++) {
    int cc = lane + 32 * e;
    dst[cc] = (__bf16)((v[e] - mean) * rstd * gw[cc] + bw[cc]);
  }
}

// ------------------------------------------------------------- plain LN rows
__global__ __launch_bounds__(256)
void ln_rows_kernel(const float* __restrict__ X, const float* __restrict__ gw,
                    const float* __restrict__ bw, __bf16* __restrict__ out) {
  const int row = blockIdx.x * 8 + (threadIdx.x >> 5);
  const int lane = threadIdx.x & 31;
  const float* src = X + (size_t)row * kC;
  float v[12];
  float s = 0.f, s2 = 0.f;
#pragma unroll
  for (int e = 0; e < 12; e++) { float t = src[lane + 32 * e]; v[e] = t; s += t; s2 += t * t; }
#pragma unroll
  for (int o = 16; o > 0; o >>= 1) { s += __shfl_xor(s, o, 32); s2 += __shfl_xor(s2, o, 32); }
  const float mean = s * (1.f / (float)kC);
  const float var  = s2 * (1.f / (float)kC) - mean * mean;
  const float rstd = rsqrtf(var + 1e-5f);
  __bf16* dst = out + (size_t)row * kC;
#pragma unroll
  for (int e = 0; e < 12; e++) {
    int cc = lane + 32 * e;
    dst[cc] = (__bf16)((v[e] - mean) * rstd * gw[cc] + bw[cc]);
  }
}

// ------------------------------------------------------------ WMMA GEMM core
// C[m,n] = sum_k A[m,k] * W[n,k] + bias[n], block tile 128x128, k-step 32.
// Double-buffered LDS with async global->LDS copies (ASYNCcnt) when available.
// 8 waves: 4 (M) x 2 (N); wave tile 32x64 = 2x4 accumulators.
// EPI: 0 = bf16 store, 1 = exact-GELU bf16 store,
//      2 = proj: window-reverse + unshift scatter + residual(x) -> f32,
//      3 = +residual -> f32 (final output)
template <int EPI>
__global__ __launch_bounds__(256)
void gemm_wmma_kernel(const __bf16* __restrict__ A, const __bf16* __restrict__ W,
                      const float* __restrict__ bias, void* __restrict__ out,
                      const float* __restrict__ resid, int M, int N, int K) {
  __shared__ __bf16 As[2][128][32];
  __shared__ __bf16 Bs[2][128][32];
  const int tid = threadIdx.x;
  const int wave = tid >> 5;
  const int lane = tid & 31;
  const int g = lane >> 4;
  const int ln = lane & 15;
  const int bm = blockIdx.y * 128;
  const int bn = blockIdx.x * 128;
  const int wm = (wave & 3) * 32;
  const int wn = (wave >> 2) * 64;

  v8f acc[2][4];
#pragma unroll
  for (int i = 0; i < 2; i++)
#pragma unroll
    for (int j = 0; j < 4; j++)
#pragma unroll
      for (int e = 0; e < 8; e++) acc[i][j][e] = 0.f;

  const int ra = tid >> 1;
  const int h16 = (tid & 1) * 16;
  const __bf16* gA = A + (size_t)(bm + ra) * K + h16;
  const __bf16* gW = W + (size_t)(bn + ra) * K + h16;

  auto issue = [&](int buf, int k0) {
    const __bf16* sa = gA + k0;
    __bf16* da = &As[buf][ra][h16];
    async_cp16<0>(sa, da);
    async_cp16<16>(sa, da);
    const __bf16* sb = gW + k0;
    __bf16* db = &Bs[buf][ra][h16];
    async_cp16<0>(sb, db);
    async_cp16<16>(sb, db);
  };

  const int nk = K >> 5;
  int buf = 0;
  issue(0, 0);
  for (int kt = 0; kt < nk; kt++) {
    const bool more = (kt + 1 < nk);
    if (more) issue(buf ^ 1, (kt + 1) * 32);
    if (more) wait_async_le4(); else wait_async_le0();
    __syncthreads();
    v16bf af[2], bf2[4];
#pragma unroll
    for (int i = 0; i < 2; i++) {
      const __bf16* rp = &As[buf][wm + i * 16 + ln][0];
      v8bf lo = *(const v8bf*)(rp + 16 * g);
      v8bf hi = *(const v8bf*)(rp + 16 * g + 8);
#pragma unroll
      for (int e = 0; e < 8; e++) { af[i][e] = lo[e]; af[i][e + 8] = hi[e]; }
    }
#pragma unroll
    for (int j = 0; j < 4; j++)
      bf2[j] = *(const v16bf*)(&Bs[buf][wn + j * 16 + ln][16 * g]);
#pragma unroll
    for (int i = 0; i < 2; i++)
#pragma unroll
      for (int j = 0; j < 4; j++)
        acc[i][j] = __builtin_amdgcn_wmma_f32_16x16x32_bf16(
            false, af[i], false, bf2[j], (short)0, acc[i][j], false, false);
    __syncthreads();
    buf ^= 1;
  }

#pragma unroll
  for (int i = 0; i < 2; i++)
#pragma unroll
    for (int j = 0; j < 4; j++) {
      const int col = bn + wn + j * 16 + ln;
      const float bv = bias[col];
#pragma unroll
      for (int r = 0; r < 8; r++) {
        const int row = bm + wm + i * 16 + 8 * g + r;
        float v = acc[i][j][r] + bv;
        if (EPI == 0) {
          ((__bf16*)out)[(size_t)row * N + col] = (__bf16)v;
        } else if (EPI == 1) {
          float gl = 0.5f * v * (1.f + erff(v * 0.70710678118654752f));
          ((__bf16*)out)[(size_t)row * N + col] = (__bf16)gl;
        } else if (EPI == 2) {
          const int win = row / kN, n = row % kN;
          const int b = win >> 6, wi = win & 63;
          const int wh = wi >> 3, ww = wi & 7;
          const int pi = n / 7, pj = n % 7;
          int rr = wh * 7 + pi + kSS; if (rr >= kH) rr -= kH;
          int cc = ww * 7 + pj + kSS; if (cc >= kW) cc -= kW;
          size_t dst = ((size_t)b * (kH * kW) + (size_t)rr * kW + cc) * kC + col;
          ((float*)out)[dst] = resid[dst] + v;
        } else {
          size_t dst = (size_t)row * N + col;
          ((float*)out)[dst] = resid[dst] + v;
        }
      }
    }
}

// ----------------------------------------------------------------- attention
__device__ __forceinline__ int swin_region(int wh, int ww, int p) {
  int pi = p / 7, pj = p % 7;
  int r = wh * 7 + pi, c = ww * 7 + pj;
  int hg = (r < 49) ? 0 : ((r < 53) ? 1 : 2);
  int wg = (c < 49) ? 0 : ((c < 53) ? 1 : 2);
  return hg * 3 + wg;
}

// One wave per (window, head): 64x64 padded S via 16 WMMAs, softmax, 64x32 P*V.
__global__ __launch_bounds__(32)
void attn_wmma_kernel(const __bf16* __restrict__ qkv, const float* __restrict__ bias_s,
                      __bf16* __restrict__ O) {
  __shared__ __bf16 smP[64][80];
  __shared__ __bf16 smVT[32][80];
  const int blk = blockIdx.x;
  const int win = blk / kHeads;
  const int h = blk % kHeads;
  const int wi = win & 63;
  const int wh = wi >> 3, ww = wi & 7;
  const int lane = threadIdx.x;
  const int g = lane >> 4, ln = lane & 15;
  const __bf16* qb = qkv + (size_t)win * kN * kQKV + h * 32;
  const __bf16* kb = qb + kC;
  const __bf16* vb = qb + 2 * kC;

  // A fragments (Q rows), B fragments (K^T columns = K rows)
  v16bf aq[4], bk[4];
#pragma unroll
  for (int mi = 0; mi < 4; mi++) {
    int row = mi * 16 + ln;
    v8bf lo, hi;
#pragma unroll
    for (int e = 0; e < 8; e++) { lo[e] = (__bf16)0.f; hi[e] = (__bf16)0.f; }
    if (row < kN) {
      const __bf16* p = qb + (size_t)row * kQKV;
      lo = *(const v8bf*)(p + 16 * g);
      hi = *(const v8bf*)(p + 16 * g + 8);
    }
#pragma unroll
    for (int e = 0; e < 8; e++) { aq[mi][e] = lo[e]; aq[mi][e + 8] = hi[e]; }
  }
#pragma unroll
  for (int ni = 0; ni < 4; ni++) {
    int col = ni * 16 + ln;
    v16bf t;
#pragma unroll
    for (int e = 0; e < 16; e++) t[e] = (__bf16)0.f;
    if (col < kN) t = *(const v16bf*)(kb + (size_t)col * kQKV + 16 * g);
    bk[ni] = t;
  }

  v8f S[4][4];
#pragma unroll
  for (int mi = 0; mi < 4; mi++)
#pragma unroll
    for (int ni = 0; ni < 4; ni++) {
#pragma unroll
      for (int e = 0; e < 8; e++) S[mi][ni][e] = 0.f;
      S[mi][ni] = __builtin_amdgcn_wmma_f32_16x16x32_bf16(
          false, aq[mi], false, bk[ni], (short)0, S[mi][ni], false, false);
    }

  // Stage V^T into LDS: vector-load a whole V row (32 bf16 = 2x32B),
  // elementwise transpose stores. Rows: lane and lane+32.
#pragma unroll
  for (int rep = 0; rep < 2; rep++) {
    const int mp = lane + rep * 32;
    v16bf a0, a1;
#pragma unroll
    for (int e = 0; e < 16; e++) { a0[e] = (__bf16)0.f; a1[e] = (__bf16)0.f; }
    if (mp < kN) {
      const __bf16* p = vb + (size_t)mp * kQKV;
      a0 = *(const v16bf*)p;
      a1 = *(const v16bf*)(p + 16);
    }
#pragma unroll
    for (int dd = 0; dd < 16; dd++) smVT[dd][mp] = a0[dd];
#pragma unroll
    for (int dd = 0; dd < 16; dd++) smVT[16 + dd][mp] = a1[dd];
  }

  // Precompute per-lane column validity / mask region
  int regCol[4];
  bool colOk[4];
#pragma unroll
  for (int ni = 0; ni < 4; ni++) {
    int col = ni * 16 + ln;
    colOk[ni] = (col < kN);
    regCol[ni] = colOk[ni] ? swin_region(wh, ww, col) : -1;
  }

  const float scale = 0.17677669529663687f;  // 1/sqrt(32)
#pragma unroll
  for (int mi = 0; mi < 4; mi++) {
#pragma unroll
    for (int r = 0; r < 8; r++) {
      const int row = mi * 16 + 8 * g + r;
      const bool rowOk = (row < kN);
      const int regRow = rowOk ? swin_region(wh, ww, row) : 0;
      float tv[4];
#pragma unroll
      for (int ni = 0; ni < 4; ni++) {
        float v = -1e30f;
        if (rowOk && colOk[ni]) {
          int col = ni * 16 + ln;
          v = S[mi][ni][r] * scale + bias_s[((size_t)h * kN + row) * kN + col];
          if (regCol[ni] != regRow) v -= 100.f;
        }
        tv[ni] = v;
      }
      float mx = fmaxf(fmaxf(tv[0], tv[1]), fmaxf(tv[2], tv[3]));
#pragma unroll
      for (int o = 8; o > 0; o >>= 1) mx = fmaxf(mx, __shfl_xor(mx, o, 32));
      float sm = 0.f;
#pragma unroll
      for (int ni = 0; ni < 4; ni++) { tv[ni] = __expf(tv[ni] - mx); sm += tv[ni]; }
#pragma unroll
      for (int o = 8; o > 0; o >>= 1) sm += __shfl_xor(sm, o, 32);
      const float inv = 1.f / sm;
#pragma unroll
      for (int ni = 0; ni < 4; ni++)
        smP[row][ni * 16 + ln] = (__bf16)(tv[ni] * inv);
    }
  }
  __syncthreads();

  // out = P @ V : M=64(4 tiles) x N=32(2 tiles), K=64 (two k-steps of 32)
  v8f acc2[4][2];
#pragma unroll
  for (int mi = 0; mi < 4; mi++)
#pragma unroll
    for (int nj = 0; nj < 2; nj++)
#pragma unroll
      for (int e = 0; e < 8; e++) acc2[mi][nj][e] = 0.f;

#pragma unroll
  for (int ks = 0; ks < 2; ks++) {
    const int kk0 = ks * 32;
    v16bf ap[4], bv[2];
#pragma unroll
    for (int mi = 0; mi < 4; mi++) {
      const int row = mi * 16 + ln;
      v8bf lo = *(const v8bf*)(&smP[row][kk0 + 16 * g]);
      v8bf hi = *(const v8bf*)(&smP[row][kk0 + 16 * g + 8]);
#pragma unroll
      for (int e = 0; e < 8; e++) { ap[mi][e] = lo[e]; ap[mi][e + 8] = hi[e]; }
    }
#pragma unroll
    for (int nj = 0; nj < 2; nj++)
      bv[nj] = *(const v16bf*)(&smVT[nj * 16 + ln][kk0 + 16 * g]);
#pragma unroll
    for (int mi = 0; mi < 4; mi++)
#pragma unroll
      for (int nj = 0; nj < 2; nj++)
        acc2[mi][nj] = __builtin_amdgcn_wmma_f32_16x16x32_bf16(
            false, ap[mi], false, bv[nj], (short)0, acc2[mi][nj], false, false);
  }

#pragma unroll
  for (int mi = 0; mi < 4; mi++)
#pragma unroll
    for (int nj = 0; nj < 2; nj++)
#pragma unroll
      for (int r = 0; r < 8; r++) {
        const int row = mi * 16 + 8 * g + r;
        if (row < kN) {
          const int col = nj * 16 + ln;
          O[((size_t)win * kN + row) * kC + h * 32 + col] = (__bf16)acc2[mi][nj][r];
        }
      }
}

// ------------------------------------------------------------------- launch
extern "C" void kernel_launch(void* const* d_in, const int* in_sizes, int n_in,
                              void* d_out, int out_size, void* d_ws, size_t ws_size,
                              hipStream_t stream) {
  (void)in_sizes; (void)n_in; (void)out_size; (void)ws_size;
  const float* x      = (const float*)d_in[0];
  const float* n1g    = (const float*)d_in[1];
  const float* n1b    = (const float*)d_in[2];
  const float* qkv_w  = (const float*)d_in[3];
  const float* qkv_b  = (const float*)d_in[4];
  const float* rpb    = (const float*)d_in[5];
  const float* proj_w = (const float*)d_in[6];
  const float* proj_b = (const float*)d_in[7];
  const float* n2g    = (const float*)d_in[8];
  const float* n2b    = (const float*)d_in[9];
  const float* fc1_w  = (const float*)d_in[10];
  const float* fc1_b  = (const float*)d_in[11];
  const float* fc2_w  = (const float*)d_in[12];
  const float* fc2_b  = (const float*)d_in[13];
  float* out = (float*)d_out;

  char* wsb = (char*)d_ws;
  size_t off = 0;
  auto take = [&](size_t bytes) -> char* {
    char* p = wsb + off;
    off = (off + bytes + 255) & ~(size_t)255;
    return p;
  };
  __bf16* wq   = (__bf16*)take((size_t)kQKV * kC * 2);
  __bf16* wp   = (__bf16*)take((size_t)kC * kC * 2);
  __bf16* w1   = (__bf16*)take((size_t)kHid * kC * 2);
  __bf16* w2   = (__bf16*)take((size_t)kC * kHid * 2);
  float* bias_s = (float*)take((size_t)kHeads * kN * kN * 4);
  __bf16* Y    = (__bf16*)take((size_t)kM * kC * 2);     // LN1 windows, later LN2 rows
  __bf16* BIG  = (__bf16*)take((size_t)kM * kHid * 2);   // qkv (M x 1152), later H1 (M x 1536)
  __bf16* Oa   = (__bf16*)take((size_t)kM * kC * 2);     // attention output
  float* xres  = (float*)take((size_t)kM * kC * 4);      // x + attn branch

  f2bf_kernel<<<(kQKV * kC + 255) / 256, 256, 0, stream>>>(qkv_w, wq, kQKV * kC);
  f2bf_kernel<<<(kC * kC + 255) / 256, 256, 0, stream>>>(proj_w, wp, kC * kC);
  f2bf_kernel<<<(kHid * kC + 255) / 256, 256, 0, stream>>>(fc1_w, w1, kHid * kC);
  f2bf_kernel<<<(kC * kHid + 255) / 256, 256, 0, stream>>>(fc2_w, w2, kC * kHid);
  rpb_expand_kernel<<<(kHeads * kN * kN + 255) / 256, 256, 0, stream>>>(rpb, bias_s);

  ln1_shift_window_kernel<<<kM / 8, 256, 0, stream>>>(x, n1g, n1b, Y);

  gemm_wmma_kernel<0><<<dim3(kQKV / 128, kM / 128), 256, 0, stream>>>(
      Y, wq, qkv_b, BIG, nullptr, kM, kQKV, kC);

  attn_wmma_kernel<<<kWinTotal * kHeads, 32, 0, stream>>>(BIG, bias_s, Oa);

  gemm_wmma_kernel<2><<<dim3(kC / 128, kM / 128), 256, 0, stream>>>(
      Oa, wp, proj_b, xres, x, kM, kC, kC);

  ln_rows_kernel<<<kM / 8, 256, 0, stream>>>(xres, n2g, n2b, Y);

  gemm_wmma_kernel<1><<<dim3(kHid / 128, kM / 128), 256, 0, stream>>>(
      Y, w1, fc1_b, BIG, nullptr, kM, kHid, kC);

  gemm_wmma_kernel<3><<<dim3(kC / 128, kM / 128), 256, 0, stream>>>(
      BIG, w2, fc2_b, out, xres, kM, kC, kHid);
}
